// PracticeNet_34445637714368
// MI455X (gfx1250) — matrix operand order repaired and checked by reading, not compiled, not run on hardware
//
#include <hip/hip_runtime.h>

#define NN 100000
#define NE 3200000
#define NG 512
#define HID 32

typedef __attribute__((ext_vector_type(16))) _Float16 v16h;
typedef __attribute__((ext_vector_type(8)))  float    v8f;

// ---------------- K0: init deg (=1 for self-loop) and pooled g (=0) ----------
__global__ void k0_init(float* __restrict__ deg, float* __restrict__ g) {
    int t = blockIdx.x * blockDim.x + threadIdx.x;
    if (t < NN) deg[t] = 1.0f;
    if (t < NG * HID) g[t] = 0.0f;
}

// ---------------- K1: degree accumulation over edge targets ------------------
__global__ void k1_deg(const int* __restrict__ col, float* __restrict__ deg) {
    int e = blockIdx.x * blockDim.x + threadIdx.x;
    if (e < NE) atomicAdd(&deg[col[e]], 1.0f);
}

// ---------------- K2: dinv = rsqrt(deg) in-place; s init = dinv^2 * x (self-loop)
__global__ void k2_dinv(float* __restrict__ deg_dinv, const float* __restrict__ x,
                        float* __restrict__ s) {
    int i = blockIdx.x * blockDim.x + threadIdx.x;
    if (i >= NN) return;
    float d  = deg_dinv[i];          // >= 1 always
    float di = rsqrtf(d);
    deg_dinv[i] = di;
    s[i] = di * di * x[i];
}

// ---------------- K3: scalar edge aggregation for layer 1 --------------------
// s[col] += dinv[row]*dinv[col]*x[row]   (32x less traffic than 32-wide agg)
__global__ void k3_edge_scalar(const int* __restrict__ row, const int* __restrict__ col,
                               const float* __restrict__ dinv, const float* __restrict__ x,
                               float* __restrict__ s) {
    int e = blockIdx.x * blockDim.x + threadIdx.x;
    if (e >= NE) return;
    int r = row[e], c = col[e];
    atomicAdd(&s[c], dinv[r] * dinv[c] * x[r]);
}

// ---------------- K4: h1 = relu(s*W1 + b1); agg2 init = dinv^2 * h1 ----------
__global__ void k4_h1(const float* __restrict__ s, const float* __restrict__ dinv,
                      const float* __restrict__ W1, const float* __restrict__ b1,
                      float* __restrict__ h1, float* __restrict__ agg2) {
    int t = blockIdx.x * blockDim.x + threadIdx.x;
    if (t >= NN * HID) return;
    int i = t >> 5;           // node (wave-uniform: blockDim multiple of 32)
    int k = t & 31;           // feature = lane
    float h = fmaxf(s[i] * W1[k] + b1[k], 0.0f);
    h1[t] = h;
    float di = dinv[i];
    agg2[t] = di * di * h;    // self-loop contribution of layer 2
}

// ---------------- K5: vector edge aggregation for layer 2 --------------------
// wave handles 32 edges: lanes load edge records, then broadcast each edge so
// all 32 lanes do coalesced 128B gather of h1[row] and coalesced 128B atomic
// add into agg2[col] (both arrays L2-resident).
__global__ void k5_edge_vec(const int* __restrict__ rowp, const int* __restrict__ colp,
                            const float* __restrict__ dinv, const float* __restrict__ h1,
                            float* __restrict__ agg2) {
    int lane = threadIdx.x & 31;
    int wave = (blockIdx.x * blockDim.x + threadIdx.x) >> 5;
    int e0 = wave * 32;
    if (e0 >= NE) return;                 // NE % 32 == 0 -> full chunks
    int e = e0 + lane;
    int r = rowp[e];
    int c = colp[e];
    float nrm = dinv[r] * dinv[c];
#pragma unroll 4
    for (int i = 0; i < 32; ++i) {
        int   rr = __shfl(r, i, 32);
        int   cc = __shfl(c, i, 32);
        float nn = __shfl(nrm, i, 32);
        float val = h1[rr * HID + lane] * nn;
        atomicAdd(&agg2[cc * HID + lane], val);
    }
}

// ---------------- K6: h2 = relu(agg2 @ W2 + b2) via WMMA, fused max-pool -----
// One wave per 16 nodes. Two v_wmma_f32_16x16x32_f16 per wave (N-blocks 0..15,
// 16..31), bias pre-loaded into the f32 accumulator, relu + sorted-batch
// run-compressed atomicMax directly into g (no h2 spill to memory).
__global__ void k6_wmma_pool(const float* __restrict__ agg2, const float* __restrict__ W2,
                             const float* __restrict__ b2, const int* __restrict__ batch,
                             float* __restrict__ g) {
    int lane  = threadIdx.x & 31;
    int wave  = (blockIdx.x * blockDim.x + threadIdx.x) >> 5;
    int node0 = wave * 16;
    if (node0 >= NN) return;              // wave-uniform: EXEC all-1 for WMMA

    // ---- A fragment: 16x32 f16, lane<16 holds row m, K={0..7,16..23}; lane>=16 K={8..15,24..31}
    int m  = lane & 15;
    int kb = (lane < 16) ? 0 : 8;
    const float* arow = agg2 + (size_t)(node0 + m) * HID;
    v16h a;
#pragma unroll
    for (int j = 0; j < 4; ++j) {
        a[2 * j]     = (_Float16)arow[kb + 2 * j];
        a[2 * j + 1] = (_Float16)arow[kb + 2 * j + 1];
        a[8 + 2 * j]     = (_Float16)arow[16 + kb + 2 * j];
        a[8 + 2 * j + 1] = (_Float16)arow[16 + kb + 2 * j + 1];
    }

    int n   = lane & 15;
    int kbB = (lane < 16) ? 0 : 8;        // B: lane<16 holds K={0..7,16..23} of col n, lane>=16 K={8..15,24..31}
    int baseRow = node0 + ((lane < 16) ? 0 : 8);

#pragma unroll
    for (int cblk = 0; cblk < 2; ++cblk) {
        int colF = cblk * 16 + n;
        v16h b;
#pragma unroll
        for (int j = 0; j < 8; ++j) {
            b[2 * j]     = (_Float16)W2[(kbB + 2 * j + (j >= 4 ? 8 : 0)) * HID + colF];
            b[2 * j + 1] = (_Float16)W2[(kbB + 2 * j + 1 + (j >= 4 ? 8 : 0)) * HID + colF];
        }
        float bias = b2[colF];
        v8f cacc;
#pragma unroll
        for (int r = 0; r < 8; ++r) cacc[r] = bias;

        v8f d = __builtin_amdgcn_wmma_f32_16x16x32_f16(
            /*neg_a=*/false, a, /*neg_b=*/false, b,
            /*c_mod=*/(short)0, cacc, /*reuse_a=*/false, /*reuse_b=*/false);

        // relu + segment-max: batch sorted -> compress runs before atomics
        int   curB   = batch[baseRow];
        float curMax = fmaxf(d[0], 0.0f);
#pragma unroll
        for (int r = 1; r < 8; ++r) {
            int   bb  = batch[baseRow + r];
            float val = fmaxf(d[r], 0.0f);
            if (bb == curB) {
                curMax = fmaxf(curMax, val);
            } else {
                atomicMax((int*)&g[curB * HID + colF], __float_as_int(curMax));
                curB = bb; curMax = val;
            }
        }
        atomicMax((int*)&g[curB * HID + colF], __float_as_int(curMax));
    }
}

// ---------------- K7: logits = g @ Wl + bl, softmax over 2 classes -----------
__global__ void k7_head(const float* __restrict__ g, const float* __restrict__ Wl,
                        const float* __restrict__ bl, float* __restrict__ out) {
    int gi = blockIdx.x * blockDim.x + threadIdx.x;
    if (gi >= NG) return;
    float l0 = bl[0], l1 = bl[1];
#pragma unroll
    for (int k = 0; k < HID; ++k) {
        float v = g[gi * HID + k];
        l0 += v * Wl[k * 2 + 0];
        l1 += v * Wl[k * 2 + 1];
    }
    float mx  = fmaxf(l0, l1);
    float e0  = __expf(l0 - mx);
    float e1  = __expf(l1 - mx);
    float inv = 1.0f / (e0 + e1);
    out[gi * 2 + 0] = e0 * inv;
    out[gi * 2 + 1] = e1 * inv;
}

extern "C" void kernel_launch(void* const* d_in, const int* in_sizes, int n_in,
                              void* d_out, int out_size, void* d_ws, size_t ws_size,
                              hipStream_t stream) {
    (void)in_sizes; (void)n_in; (void)out_size; (void)ws_size;
    const float* x     = (const float*)d_in[0];
    const int*   ei    = (const int*)  d_in[1];
    const int*   batch = (const int*)  d_in[2];
    const float* W1    = (const float*)d_in[3];
    const float* b1    = (const float*)d_in[4];
    const float* W2    = (const float*)d_in[5];
    const float* b2    = (const float*)d_in[6];
    const float* Wl    = (const float*)d_in[7];
    const float* bl    = (const float*)d_in[8];
    float*       out   = (float*)d_out;

    const int* row = ei;            // edge_index[0] = source j
    const int* col = ei + NE;       // edge_index[1] = target i

    // workspace layout (floats), all buffers written before read each call
    float* ws = (float*)d_ws;
    size_t o = 0;
    float* deg_dinv = ws + o; o += 100096;               // N padded
    float* s        = ws + o; o += 100096;
    float* h1       = ws + o; o += (size_t)NN * HID;     // 12.8 MB
    float* agg2     = ws + o; o += (size_t)NN * HID;     // 12.8 MB
    float* g        = ws + o; o += (size_t)NG * HID;

    k0_init       <<<(NN + 255) / 256, 256, 0, stream>>>(deg_dinv, g);
    k1_deg        <<<(NE + 255) / 256, 256, 0, stream>>>(col, deg_dinv);
    k2_dinv       <<<(NN + 255) / 256, 256, 0, stream>>>(deg_dinv, x, s);
    k3_edge_scalar<<<(NE + 255) / 256, 256, 0, stream>>>(row, col, deg_dinv, x, s);
    k4_h1         <<<(NN * HID + 255) / 256, 256, 0, stream>>>(s, deg_dinv, W1, b1, h1, agg2);
    k5_edge_vec   <<<(NE * 32 / 32 + 255) / 256, 256, 0, stream>>>(row, col, deg_dinv, h1, agg2); // 100000 waves
    k6_wmma_pool  <<<((NN / 16) * 32 + 255) / 256, 256, 0, stream>>>(agg2, W2, b2, batch, g);     // 6250 waves
    k7_head       <<<2, 256, 0, stream>>>(g, Wl, bl, out);
}
